// FullAttention_63196148794096
// MI455X (gfx1250) — matrix operand order repaired and checked by reading, not compiled, not run on hardware
//
#include <hip/hip_runtime.h>

typedef __attribute__((ext_vector_type(16))) _Float16 v16h;
typedef __attribute__((ext_vector_type(8)))  _Float16 v8h;
typedef __attribute__((ext_vector_type(2)))  _Float16 v2h;
typedef __attribute__((ext_vector_type(8)))  float    v8f;
typedef __attribute__((ext_vector_type(8)))  unsigned int v8u;

constexpr int   Bb = 2, Ll = 2048, Hh = 16, Ee = 64;
constexpr int   RS = Hh * Ee;                               // 1024 floats between seq positions
constexpr float QSCALE = 0.125f * 1.4426950408889634f;      // (1/sqrt(E)) * log2(e): exp2 domain

#define WMMA_F16(A, Bm, C) \
  __builtin_amdgcn_wmma_f32_16x16x32_f16(false, (A), false, (Bm), (short)0, (C), false, false)

static __device__ __forceinline__ v16h cat8(v8h lo, v8h hi) {
  v16h r;
#pragma unroll
  for (int i = 0; i < 8; ++i) { r[i] = lo[i]; r[8 + i] = hi[i]; }
  return r;
}

__global__ __launch_bounds__(128, 1)
void fa_causal_fwd(const float* __restrict__ Q, const float* __restrict__ K,
                   const float* __restrict__ V, float* __restrict__ O)
{
  // double-buffered cooperative K (row-major) / V (transposed) f16 tiles
  __shared__ __align__(32) _Float16 kbuf[2][32][64];   // [buf][s][e]   8 KB
  __shared__ __align__(32) _Float16 vbuf[2][64][32];   // [buf][d][s]   8 KB

  const int tid  = threadIdx.x;
  const int lane = tid & 31;
  const int wave = tid >> 5;
  const int ln   = lane & 15;          // query column m owned by this lane (S^T layout)
  const int hs   = lane >> 4;          // half-wave select

  const int qblk = blockIdx.x;
  const int h    = blockIdx.y;
  const int b    = blockIdx.z;

  const int    q0   = qblk * 64 + wave * 16;
  const size_t base = (size_t)b * Ll * RS + (size_t)h * Ee;
  const float  NEG_INF = -__builtin_huge_valf();

  // ---- Q B-fragments (column m = ln; K-halves = e contiguous), scale folded ----
  v16h bq0, bq1;
  {
    const float4* q4 = (const float4*)(Q + base + (size_t)(q0 + ln) * RS);
#pragma unroll
    for (int c = 0; c < 4; ++c) {
      float4 t = q4[hs * 4 + c];                 // e = hs*16 + 4c ..
      bq0[4*c+0]=(_Float16)(t.x*QSCALE); bq0[4*c+1]=(_Float16)(t.y*QSCALE);
      bq0[4*c+2]=(_Float16)(t.z*QSCALE); bq0[4*c+3]=(_Float16)(t.w*QSCALE);
      float4 u = q4[8 + hs * 4 + c];             // e = 32 + hs*16 + 4c ..
      bq1[4*c+0]=(_Float16)(u.x*QSCALE); bq1[4*c+1]=(_Float16)(u.y*QSCALE);
      bq1[4*c+2]=(_Float16)(u.z*QSCALE); bq1[4*c+3]=(_Float16)(u.w*QSCALE);
    }
  }

  // all-ones A fragment: denominator = 1^T x P^T rides the WMMA pipe
  v16h aone;
#pragma unroll
  for (int i = 0; i < 16; ++i) aone[i] = (_Float16)1.0f;

  v8f acc0 = {}, acc1 = {}, acc2 = {}, acc3 = {};  // O^T accumulators, d-tiles 0..3
  v8f accs = {};                                   // softmax denominator
  float m_run = NEG_INF;                           // ONE running max per lane (= per query col)

  const int n_iter = qblk * 2 + 2;                 // block-uniform: keys 0..qblk*64+63

  // staging maps: K = 4 threads/row x 16 floats; V = (d-chunk, s-pair) for packed stores
  const int srow = tid >> 2, quad = tid & 3;
  const int dq   = tid & 7,  sp   = tid >> 3;      // dq: 8 d-chunks, sp: 16 s-pairs
  const float* kg = K + base + (size_t)srow * RS + quad * 16;
  const float* vg = V + base + (size_t)(2 * sp) * RS + dq * 8;

  float4 kr[4], vr[4];
#pragma unroll
  for (int c = 0; c < 4; ++c) kr[c] = ((const float4*)kg)[c];
  vr[0] = ((const float4*)vg)[0];        vr[1] = ((const float4*)vg)[1];
  vr[2] = ((const float4*)(vg + RS))[0]; vr[3] = ((const float4*)(vg + RS))[1];

  for (int it = 0; it < n_iter; ++it) {
    const int S0  = it * 32;
    const int buf = it & 1;

    // ---- stage K tile row-major f16 (2x b128 LDS stores) ----
    {
      v8h h0, h1;
      h0[0]=(_Float16)kr[0].x; h0[1]=(_Float16)kr[0].y; h0[2]=(_Float16)kr[0].z; h0[3]=(_Float16)kr[0].w;
      h0[4]=(_Float16)kr[1].x; h0[5]=(_Float16)kr[1].y; h0[6]=(_Float16)kr[1].z; h0[7]=(_Float16)kr[1].w;
      h1[0]=(_Float16)kr[2].x; h1[1]=(_Float16)kr[2].y; h1[2]=(_Float16)kr[2].z; h1[3]=(_Float16)kr[2].w;
      h1[4]=(_Float16)kr[3].x; h1[5]=(_Float16)kr[3].y; h1[6]=(_Float16)kr[3].z; h1[7]=(_Float16)kr[3].w;
      *(v8h*)&kbuf[buf][srow][quad * 16]     = h0;
      *(v8h*)&kbuf[buf][srow][quad * 16 + 8] = h1;
    }
    // ---- stage V tile transposed [d][s], packed s-pair b32 stores ----
    {
      const float* r0 = (const float*)&vr[0];    // 8 floats of row 2*sp
      const float* r1 = (const float*)&vr[2];    // 8 floats of row 2*sp+1
#pragma unroll
      for (int i = 0; i < 8; ++i) {
        v2h pp; pp[0] = (_Float16)r0[i]; pp[1] = (_Float16)r1[i];
        *(v2h*)&vbuf[buf][dq * 8 + i][2 * sp] = pp;
      }
    }
    // ---- register-prefetch next tile; L2-prefetch the one after ----
    if (it + 1 < n_iter) {
      const float4* kn = (const float4*)(kg + (size_t)(it + 1) * 32 * RS);
      const float*  vn = vg + (size_t)(it + 1) * 32 * RS;
#pragma unroll
      for (int c = 0; c < 4; ++c) kr[c] = kn[c];
      vr[0] = ((const float4*)vn)[0];        vr[1] = ((const float4*)vn)[1];
      vr[2] = ((const float4*)(vn + RS))[0]; vr[3] = ((const float4*)(vn + RS))[1];
    }
    if (it + 2 < n_iter) {
      __builtin_prefetch(kg + (size_t)(it + 2) * 32 * RS, 0, 1);
      __builtin_prefetch(vg + (size_t)(it + 2) * 32 * RS, 0, 1);
    }
    __syncthreads();   // stage(it) visible; double buffer fences reuse

    // ---- S^T = K x Q^T : A-fragments of K straight from LDS ----
    v16h ak00 = cat8(*(const v8h*)&kbuf[buf][ln     ][     hs*16], *(const v8h*)&kbuf[buf][ln     ][ 8 + hs*16]);
    v16h ak01 = cat8(*(const v8h*)&kbuf[buf][ln     ][32 + hs*16], *(const v8h*)&kbuf[buf][ln     ][40 + hs*16]);
    v16h ak10 = cat8(*(const v8h*)&kbuf[buf][16 + ln][     hs*16], *(const v8h*)&kbuf[buf][16 + ln][ 8 + hs*16]);
    v16h ak11 = cat8(*(const v8h*)&kbuf[buf][16 + ln][32 + hs*16], *(const v8h*)&kbuf[buf][16 + ln][40 + hs*16]);

    v8f c0 = {}, c1 = {};                      // S^T tiles: lane = query col, regs = key rows
    c0 = WMMA_F16(ak00, bq0, c0);
    c0 = WMMA_F16(ak01, bq1, c0);
    c1 = WMMA_F16(ak10, bq0, c1);
    c1 = WMMA_F16(ak11, bq1, c1);

    // causal mask (wave-uniform guard): s = S0 + t*16 + hs*8 + r vs q = q0 + ln
    if (S0 + 31 > q0) {
      const int qq = q0 + ln;
#pragma unroll
      for (int r = 0; r < 8; ++r) {
        c0[r] = (S0 +      hs*8 + r > qq) ? NEG_INF : c0[r];
        c1[r] = (S0 + 16 + hs*8 + r > qq) ? NEG_INF : c1[r];
      }
    }

    // ---- V^T A-fragments from transposed LDS tile ----
    v16h av0 = cat8(*(const v8h*)&vbuf[buf][ln     ][hs*16], *(const v8h*)&vbuf[buf][ln     ][8 + hs*16]);
    v16h av1 = cat8(*(const v8h*)&vbuf[buf][16 + ln][hs*16], *(const v8h*)&vbuf[buf][16 + ln][8 + hs*16]);
    v16h av2 = cat8(*(const v8h*)&vbuf[buf][32 + ln][hs*16], *(const v8h*)&vbuf[buf][32 + ln][8 + hs*16]);
    v16h av3 = cat8(*(const v8h*)&vbuf[buf][48 + ln][hs*16], *(const v8h*)&vbuf[buf][48 + ln][8 + hs*16]);

    // ---- online softmax: in-lane max over 16 regs + ONE partner shuffle ----
    float mx = fmaxf(c0[0], c1[0]);
#pragma unroll
    for (int r = 1; r < 8; ++r) mx = fmaxf(mx, fmaxf(c0[r], c1[r]));
    mx = fmaxf(mx, __shfl_xor(mx, 16, 32));
    const float mnew  = fmaxf(m_run, mx);
    const float alpha = __builtin_amdgcn_exp2f(m_run - mnew);
    m_run = mnew;

    // P^T values (f16) for this lane's column; then one xor-16 exchange builds the
    // B fragment in the k order of av0..av3 (s = hs*16 + i contiguous) -- no LDS round trip.
    v16h own;
#pragma unroll
    for (int r = 0; r < 8; ++r) {
      own[r]     = (_Float16)__builtin_amdgcn_exp2f(c0[r] - mnew);   // s = S0 +      hs*8 + r
      own[8 + r] = (_Float16)__builtin_amdgcn_exp2f(c1[r] - mnew);   // s = S0 + 16 + hs*8 + r
    }
    v8u u = __builtin_bit_cast(v8u, own), pu;
#pragma unroll
    for (int j = 0; j < 8; ++j) pu[j] = (unsigned)__shfl_xor((int)u[j], 16, 32);
    v8u selv;
#pragma unroll
    for (int j = 0; j < 4; ++j) {
      selv[j]     = hs ? pu[4 + j] : u[j];       // halves 0..7 : s = S0 + hs*16 + 0..7
      selv[4 + j] = hs ? u[4 + j]  : pu[j];      // halves 8..15: s = S0 + hs*16 + 8..15
    }
    v16h bpt = __builtin_bit_cast(v16h, selv);

    // rescale accumulators by lane-scalar alpha
#pragma unroll
    for (int r = 0; r < 8; ++r) {
      acc0[r] *= alpha; acc1[r] *= alpha; acc2[r] *= alpha; acc3[r] *= alpha; accs[r] *= alpha;
    }

    // ---- O^T += V^T x P^T (4 d-tiles) ; denominator += 1^T x P^T ----
    acc0 = WMMA_F16(av0, bpt, acc0);
    acc1 = WMMA_F16(av1, bpt, acc1);
    acc2 = WMMA_F16(av2, bpt, acc2);
    acc3 = WMMA_F16(av3, bpt, acc3);
    accs = WMMA_F16(aone, bpt, accs);
  }

  // ---- epilogue: O^T layout gives 8 consecutive d per lane -> b128 stores ----
  const float inv = 1.0f / accs[0];              // all rows of accs are identical
  float* orow = O + base + (size_t)(q0 + ln) * RS;
  {
    float4 s0, s1;
    s0.x=acc0[0]*inv; s0.y=acc0[1]*inv; s0.z=acc0[2]*inv; s0.w=acc0[3]*inv;
    s1.x=acc0[4]*inv; s1.y=acc0[5]*inv; s1.z=acc0[6]*inv; s1.w=acc0[7]*inv;
    *(float4*)(orow +  0 + hs*8) = s0; *(float4*)(orow +  0 + hs*8 + 4) = s1;
    s0.x=acc1[0]*inv; s0.y=acc1[1]*inv; s0.z=acc1[2]*inv; s0.w=acc1[3]*inv;
    s1.x=acc1[4]*inv; s1.y=acc1[5]*inv; s1.z=acc1[6]*inv; s1.w=acc1[7]*inv;
    *(float4*)(orow + 16 + hs*8) = s0; *(float4*)(orow + 16 + hs*8 + 4) = s1;
    s0.x=acc2[0]*inv; s0.y=acc2[1]*inv; s0.z=acc2[2]*inv; s0.w=acc2[3]*inv;
    s1.x=acc2[4]*inv; s1.y=acc2[5]*inv; s1.z=acc2[6]*inv; s1.w=acc2[7]*inv;
    *(float4*)(orow + 32 + hs*8) = s0; *(float4*)(orow + 32 + hs*8 + 4) = s1;
    s0.x=acc3[0]*inv; s0.y=acc3[1]*inv; s0.z=acc3[2]*inv; s0.w=acc3[3]*inv;
    s1.x=acc3[4]*inv; s1.y=acc3[5]*inv; s1.z=acc3[6]*inv; s1.w=acc3[7]*inv;
    *(float4*)(orow + 48 + hs*8) = s0; *(float4*)(orow + 48 + hs*8 + 4) = s1;
  }
}

extern "C" void kernel_launch(void* const* d_in, const int* in_sizes, int n_in,
                              void* d_out, int out_size, void* d_ws, size_t ws_size,
                              hipStream_t stream) {
  (void)in_sizes; (void)n_in; (void)d_ws; (void)ws_size; (void)out_size;
  const float* Q = (const float*)d_in[0];
  const float* K = (const float*)d_in[1];
  const float* V = (const float*)d_in[2];
  float* O = (float*)d_out;

  dim3 grid(Ll / 64, Hh, Bb);   // (q-tile, head, batch)
  dim3 block(128);              // 4 waves, one 16-query strip each
  fa_causal_fwd<<<grid, block, 0, stream>>>(Q, K, V, O);
}
